// SpaMamba_91036126806277
// MI455X (gfx1250) — matrix operand order repaired and checked by reading, not compiled, not run on hardware
//
#include <hip/hip_runtime.h>
#include <hip/hip_bf16.h>

// ---------------------------------------------------------------------------
// SpaMamba fused pipeline for gfx1250 (MI455X), wave32 + WMMA bf16.
// ---------------------------------------------------------------------------

typedef __attribute__((ext_vector_type(16))) __bf16 v16bf;
typedef __attribute__((ext_vector_type(8)))  float  v8f;

#define D_MODEL 128
#define D_INNER 256
#define D_STATE 16
#define DT_RANK 8
#define NB      4
#define LSEQ    16384
#define NTOK    (NB * LSEQ)  // 65536 tokens

// bf16 weight blob element offsets
#define OFF_W1   0         // mlp_w1   256x128
#define OFF_W2   32768     // mlp_w2   128x256
#define OFF_WIN  65536     // in_proj  512x128
#define OFF_WXP  131072    // x_proj   48x256 (padded from 40 rows)
#define OFF_WDT  143360    // dt_proj  256x8
#define OFF_WOUT 145408    // out_proj 128x256
#define NBLOB    178176

union BU { v16bf v; uint4 u[2]; };

// B fragment (32x16 KxN) from row-major W[out][in]: B[k][n] = W[n][k].
// Lane: n = lane&15, half = lane>>4; e<8 -> K=k0+8h+e ; e>=8 -> K=k0+16+8h+(e-8)
__device__ __forceinline__ v16bf ld_bfrag(const __bf16* W, int ldw, int n0, int k0, int lane) {
  int col = lane & 15, half = lane >> 4;
  const __bf16* p = W + (size_t)(n0 + col) * ldw + (k0 + 8 * half);
  BU r; r.u[0] = *(const uint4*)p; r.u[1] = *(const uint4*)(p + 16);
  return r.v;
}

// A fragment (16x32 MxK) from row-major bf16 source.
// Lane: m = lane&15, half = lane>>4; e<8 -> K=k0+8h+e ; e>=8 -> K=k0+16+8h+(e-8)
__device__ __forceinline__ v16bf ld_afrag(const __bf16* A, int lda, int m0, int k0, int lane) {
  int row = lane & 15, half = lane >> 4;
  const __bf16* p = A + (size_t)(m0 + row) * lda + (k0 + 8 * half);
  BU r; r.u[0] = *(const uint4*)p; r.u[1] = *(const uint4*)(p + 16);
  return r.v;
}

__device__ __forceinline__ v8f wmma_bf(v16bf a, v16bf b, v8f c) {
  return __builtin_amdgcn_wmma_f32_16x16x32_bf16(false, a, false, b, (short)0, c, false, false);
}

// ---------------------------------------------------------------------------
// Kernel 0: convert all weights to one bf16 blob (pads x_proj to 48 rows),
//           also zero the GroupNorm stats accumulator.
// ---------------------------------------------------------------------------
__global__ __launch_bounds__(256) void k_convert(
    const float* __restrict__ w1, const float* __restrict__ w2,
    const float* __restrict__ win, const float* __restrict__ wxp,
    const float* __restrict__ wdt, const float* __restrict__ wout,
    __bf16* __restrict__ blob, float* __restrict__ stats)
{
  int idx = blockIdx.x * 256 + threadIdx.x;
  if (idx < 32) stats[idx] = 0.f;
  if (idx >= NBLOB) return;
  float v;
  if (idx < OFF_W2)       v = w1[idx];
  else if (idx < OFF_WIN) v = w2[idx - OFF_W2];
  else if (idx < OFF_WXP) v = win[idx - OFF_WIN];
  else if (idx < OFF_WDT) {
    int l = idx - OFF_WXP;
    v = ((l >> 8) < 40) ? wxp[l] : 0.f;   // zero-pad rows 40..47
  }
  else if (idx < OFF_WOUT) v = wdt[idx - OFF_WDT];
  else                     v = wout[idx - OFF_WOUT];
  blob[idx] = (__bf16)v;
}

// ---------------------------------------------------------------------------
// Kernel 1: fused  xm = relu(xf@W1^T+b1)@W2^T+b2 ;  [x_in | z] = xm@Win^T
// One wave per 16-token tile; intermediates staged per-wave in LDS (bf16).
// ---------------------------------------------------------------------------
__global__ __launch_bounds__(128) void k_mlp_inproj(
    const float* __restrict__ x, const __bf16* __restrict__ blob,
    const float* __restrict__ b1, const float* __restrict__ b2,
    __bf16* __restrict__ x_in, __bf16* __restrict__ z)
{
  __shared__ __align__(16) __bf16 s_h1[4][16 * 256];
  __shared__ __align__(16) __bf16 s_xm[4][16 * 128];
  int wave = threadIdx.x >> 5, lane = threadIdx.x & 31;
  int tile = blockIdx.x * 4 + wave;
  int m0 = tile << 4;
  int row = lane & 15, half = lane >> 4, col = row;

  // A fragments straight out of NCHW x: token n=(b,t), K = channel c (stride H*W)
  int n = m0 + row;
  int bb = n >> 14, t = n & (LSEQ - 1);
  const float* xb = x + (((size_t)bb * D_MODEL) << 14) + t;
  v16bf a_x[4];
  #pragma unroll
  for (int kf = 0; kf < 4; ++kf) {
    v16bf a;
    #pragma unroll
    for (int e = 0; e < 16; ++e) {
      int K = kf * 32 + ((e < 8) ? (8 * half + e) : (8 + 8 * half + e));
      a[e] = (__bf16)xb[(size_t)K << 14];
    }
    a_x[kf] = a;
  }

  // GEMM1: h1 = relu(xf @ W1^T + b1)  (16x256)
  const __bf16* w1 = blob + OFF_W1;
  __bf16* h1p = &s_h1[wave][0];
  #pragma unroll 1
  for (int nt = 0; nt < 16; ++nt) {
    v8f acc = {};
    #pragma unroll
    for (int kf = 0; kf < 4; ++kf)
      acc = wmma_bf(a_x[kf], ld_bfrag(w1, 128, nt * 16, kf * 32, lane), acc);
    float bv = b1[nt * 16 + col];
    #pragma unroll
    for (int r = 0; r < 8; ++r) {
      float v = fmaxf(acc[r] + bv, 0.f);
      h1p[(r + 8 * half) * 256 + nt * 16 + col] = (__bf16)v;
    }
  }

  // GEMM2: xm = h1 @ W2^T + b2  (16x128)
  v16bf a_h[8];
  #pragma unroll
  for (int kf = 0; kf < 8; ++kf) a_h[kf] = ld_afrag(h1p, 256, 0, kf * 32, lane);
  const __bf16* w2 = blob + OFF_W2;
  __bf16* xmp = &s_xm[wave][0];
  #pragma unroll 1
  for (int nt = 0; nt < 8; ++nt) {
    v8f acc = {};
    #pragma unroll
    for (int kf = 0; kf < 8; ++kf)
      acc = wmma_bf(a_h[kf], ld_bfrag(w2, 256, nt * 16, kf * 32, lane), acc);
    float bv = b2[nt * 16 + col];
    #pragma unroll
    for (int r = 0; r < 8; ++r)
      xmp[(r + 8 * half) * 128 + nt * 16 + col] = (__bf16)(acc[r] + bv);
  }

  // GEMM3: [x_in | z] = xm @ Win^T  (16x512)
  v16bf a_m[4];
  #pragma unroll
  for (int kf = 0; kf < 4; ++kf) a_m[kf] = ld_afrag(xmp, 128, 0, kf * 32, lane);
  const __bf16* win = blob + OFF_WIN;
  #pragma unroll 1
  for (int nt = 0; nt < 32; ++nt) {
    v8f acc = {};
    #pragma unroll
    for (int kf = 0; kf < 4; ++kf)
      acc = wmma_bf(a_m[kf], ld_bfrag(win, 128, nt * 16, kf * 32, lane), acc);
    __bf16* dst = (nt < 16) ? x_in : z;
    int cg = (nt & 15) * 16 + col;
    #pragma unroll
    for (int r = 0; r < 8; ++r)
      dst[(size_t)(m0 + r + 8 * half) * 256 + cg] = (__bf16)acc[r];
  }
}

// ---------------------------------------------------------------------------
// Kernel 2: causal depthwise conv1d (k=4) + bias + SiLU  -> u (bf16)
// ---------------------------------------------------------------------------
__global__ __launch_bounds__(256) void k_conv(
    const __bf16* __restrict__ x_in, const float* __restrict__ cw,
    const float* __restrict__ cb, __bf16* __restrict__ u)
{
  size_t idx = (size_t)blockIdx.x * 256 + threadIdx.x;  // NTOK*256
  int d = (int)(idx & 255);
  size_t nn = idx >> 8;
  int t = (int)(nn & (LSEQ - 1));
  float acc = cb[d];
  const float* w = cw + d * 4;
  #pragma unroll
  for (int j = 0; j < 4; ++j) {
    int tt = t - 3 + j;
    if (tt >= 0) acc += w[j] * (float)x_in[idx - (size_t)(3 - j) * 256];
  }
  float sg = 1.f / (1.f + __expf(-acc));
  u[idx] = (__bf16)(acc * sg);
}

// ---------------------------------------------------------------------------
// Kernel 3: x_dbl = u @ Wxp^T ; split dt/B/C ; delta = softplus(dt@Wdt^T + b)
// ---------------------------------------------------------------------------
__global__ __launch_bounds__(128) void k_xproj(
    const __bf16* __restrict__ u, const __bf16* __restrict__ blob,
    const float* __restrict__ dtb, float* __restrict__ BC,
    __bf16* __restrict__ delta)
{
  __shared__ __align__(16) __bf16 s_dt[4][16 * 32];
  int wave = threadIdx.x >> 5, lane = threadIdx.x & 31;
  int tile = blockIdx.x * 4 + wave;
  int m0 = tile << 4;
  int col = lane & 15, half = lane >> 4;
  __bf16* dtp = &s_dt[wave][0];
  for (int i = lane; i < 16 * 32; i += 32) dtp[i] = (__bf16)0.f;

  v16bf a_u[8];
  #pragma unroll
  for (int kf = 0; kf < 8; ++kf) a_u[kf] = ld_afrag(u, 256, m0, kf * 32, lane);

  const __bf16* wxp = blob + OFF_WXP;
  #pragma unroll 1
  for (int nt = 0; nt < 3; ++nt) {              // 48 padded output cols
    v8f acc = {};
    #pragma unroll
    for (int kf = 0; kf < 8; ++kf)
      acc = wmma_bf(a_u[kf], ld_bfrag(wxp, 256, nt * 16, kf * 32, lane), acc);
    int cg = nt * 16 + col;
    #pragma unroll
    for (int r = 0; r < 8; ++r) {
      int mr = r + 8 * half;
      if (cg < DT_RANK)
        dtp[mr * 32 + cg] = (__bf16)acc[r];
      else if (cg < DT_RANK + 2 * D_STATE)
        BC[(size_t)(m0 + mr) * 32 + (cg - DT_RANK)] = acc[r];
    }
  }

  // delta = softplus(dt @ Wdt^T + dtb): A is 16x32 with K>=8 zero-padded.
  v16bf a_dt = ld_afrag(dtp, 32, 0, 0, lane);
  const __bf16* wdt = blob + OFF_WDT;
  #pragma unroll 1
  for (int nt = 0; nt < 16; ++nt) {
    v16bf bf;
    #pragma unroll
    for (int e = 0; e < 16; ++e) {
      int K = (e < 8) ? (8 * half + e) : (16 + 8 * half + (e - 8));
      bf[e] = (K < DT_RANK) ? wdt[(nt * 16 + col) * DT_RANK + K] : (__bf16)0.f;
    }
    v8f acc = {};
    acc = wmma_bf(a_dt, bf, acc);
    float bv = dtb[nt * 16 + col];
    #pragma unroll
    for (int r = 0; r < 8; ++r) {
      float v = acc[r] + bv;
      float sp = (v > 20.f) ? v : __logf(1.f + __expf(v));
      delta[(size_t)(m0 + r + 8 * half) * 256 + nt * 16 + col] = (__bf16)sp;
    }
  }
}

// ---------------------------------------------------------------------------
// Kernel 4: selective scan, 1024 independent (b,d) recurrences, 16 states in
// registers. B/C staged in LDS per 64-step chunk; prefetch next chunk.
// Fused epilogue: y = (scan + u*D) * silu(z)  -> bf16.
// ---------------------------------------------------------------------------
__global__ __launch_bounds__(256) void k_scan(
    const __bf16* __restrict__ delta, const __bf16* __restrict__ u,
    const float* __restrict__ BC, const __bf16* __restrict__ z,
    const float* __restrict__ A_log, const float* __restrict__ Dw,
    __bf16* __restrict__ y)
{
  int b = blockIdx.x, d = threadIdx.x;
  float A[16];
  #pragma unroll
  for (int s = 0; s < 16; ++s) A[s] = -__expf(A_log[d * 16 + s]);
  float Dd = Dw[d];
  float h[16];
  #pragma unroll
  for (int s = 0; s < 16; ++s) h[s] = 0.f;

  __shared__ __align__(16) float sBC[64 * 32];
  const float* BCb = BC + (((size_t)b) << 14) * 32;

  for (int t0 = 0; t0 < LSEQ; t0 += 64) {
    __syncthreads();
    #pragma unroll
    for (int i = 0; i < 8; ++i) sBC[i * 256 + d] = BCb[(size_t)t0 * 32 + i * 256 + d];
    if (t0 + 64 < LSEQ)
      __builtin_prefetch(&BCb[(size_t)(t0 + 64) * 32 + d * 8], 0, 1);
    __syncthreads();
    for (int tt = 0; tt < 64; ++tt) {
      size_t base = ((((size_t)b) << 14) + t0 + tt) * 256 + d;
      float dl = (float)delta[base];
      float ut = (float)u[base];
      const float* Bt = &sBC[tt * 32];
      float yv = 0.f;
      #pragma unroll
      for (int s = 0; s < 16; ++s) {
        float dA = __expf(dl * A[s]);
        float hs = h[s] * dA + (dl * Bt[s]) * ut;
        h[s] = hs;
        yv += hs * Bt[16 + s];
      }
      float zv = (float)z[base];
      float sg = 1.f / (1.f + __expf(-zv));
      yv = (yv + ut * Dd) * (zv * sg);
      y[base] = (__bf16)yv;
    }
  }
}

// ---------------------------------------------------------------------------
// Kernel 5: out = y @ Wout^T (f32) + per-(batch,group) sum/sumsq partials
// (wave32 shuffle reduction, one atomicAdd pair per wave per group).
// ---------------------------------------------------------------------------
__global__ __launch_bounds__(128) void k_outproj(
    const __bf16* __restrict__ y, const __bf16* __restrict__ blob,
    float* __restrict__ out, float* __restrict__ stats)
{
  int wave = threadIdx.x >> 5, lane = threadIdx.x & 31;
  int tile = blockIdx.x * 4 + wave;
  int m0 = tile << 4;
  int col = lane & 15, half = lane >> 4;
  int bb = m0 >> 14;

  v16bf a_y[8];
  #pragma unroll
  for (int kf = 0; kf < 8; ++kf) a_y[kf] = ld_afrag(y, 256, m0, kf * 32, lane);

  const __bf16* wo = blob + OFF_WOUT;
  float gs[4] = {0.f, 0.f, 0.f, 0.f}, gq[4] = {0.f, 0.f, 0.f, 0.f};
  #pragma unroll 1
  for (int nt = 0; nt < 8; ++nt) {
    v8f acc = {};
    #pragma unroll
    for (int kf = 0; kf < 8; ++kf)
      acc = wmma_bf(a_y[kf], ld_bfrag(wo, 256, nt * 16, kf * 32, lane), acc);
    int g = nt >> 1;  // 32-channel groups, 16-col tiles
    float s = 0.f, q = 0.f;
    #pragma unroll
    for (int r = 0; r < 8; ++r) {
      float v = acc[r];
      out[(size_t)(m0 + r + 8 * half) * 128 + nt * 16 + col] = v;
      s += v; q += v * v;
    }
    gs[g] += s; gq[g] += q;
  }
  #pragma unroll
  for (int g = 0; g < 4; ++g) {
    float s = gs[g], q = gq[g];
    #pragma unroll
    for (int off = 16; off > 0; off >>= 1) {
      s += __shfl_xor(s, off, 32);
      q += __shfl_xor(q, off, 32);
    }
    if (lane == 0) {
      atomicAdd(&stats[(bb * 4 + g) * 2 + 0], s);
      atomicAdd(&stats[(bb * 4 + g) * 2 + 1], q);
    }
  }
}

// ---------------------------------------------------------------------------
// Kernel 6: finalize GroupNorm stats -> (mean, rsqrt(var+eps)) per (b,group)
// ---------------------------------------------------------------------------
__global__ void k_gnstats(const float* __restrict__ stats, float* __restrict__ musig) {
  int i = threadIdx.x;
  if (i < 16) {
    const float cnt = 32.f * 16384.f;   // (C/G)*H*W
    float mean = stats[i * 2] / cnt;
    float var  = stats[i * 2 + 1] / cnt - mean * mean;
    musig[i * 2]     = mean;
    musig[i * 2 + 1] = rsqrtf(var + 1e-5f);
  }
}

// ---------------------------------------------------------------------------
// Kernel 7: GroupNorm apply + SiLU + residual, (B,L,C) -> (B,C,H,W)
// ---------------------------------------------------------------------------
__global__ __launch_bounds__(256) void k_final(
    const float* __restrict__ out, const float* __restrict__ musig,
    const float* __restrict__ gw, const float* __restrict__ gb,
    const float* __restrict__ x, float* __restrict__ o)
{
  size_t idx = (size_t)blockIdx.x * 256 + threadIdx.x;  // B*C*H*W
  int hw = (int)(idx & (LSEQ - 1));
  int c  = (int)((idx >> 14) & (D_MODEL - 1));
  int bb = (int)(idx >> 21);
  int g = c >> 5;
  float v  = out[((((size_t)bb) << 14) + hw) * 128 + c];
  float mu = musig[(bb * 4 + g) * 2];
  float rs = musig[(bb * 4 + g) * 2 + 1];
  float nv = (v - mu) * rs * gw[c] + gb[c];
  float sg = 1.f / (1.f + __expf(-nv));
  o[idx] = nv * sg + x[idx];
}

// ---------------------------------------------------------------------------
extern "C" void kernel_launch(void* const* d_in, const int* in_sizes, int n_in,
                              void* d_out, int out_size, void* d_ws, size_t ws_size,
                              hipStream_t stream)
{
  (void)in_sizes; (void)n_in; (void)out_size; (void)ws_size;
  const float* x      = (const float*)d_in[0];
  const float* mlp_w1 = (const float*)d_in[1];
  const float* mlp_b1 = (const float*)d_in[2];
  const float* mlp_w2 = (const float*)d_in[3];
  const float* mlp_b2 = (const float*)d_in[4];
  const float* in_w   = (const float*)d_in[5];
  const float* conv_w = (const float*)d_in[6];
  const float* conv_b = (const float*)d_in[7];
  const float* xp_w   = (const float*)d_in[8];
  const float* dt_w   = (const float*)d_in[9];
  const float* dt_b   = (const float*)d_in[10];
  const float* A_log  = (const float*)d_in[11];
  const float* Dw     = (const float*)d_in[12];
  const float* out_w  = (const float*)d_in[13];
  const float* gn_w   = (const float*)d_in[14];
  const float* gn_b   = (const float*)d_in[15];

  char* ws = (char*)d_ws;
  const size_t BUF = 33554432ull;   // NTOK*256*2 bytes (bf16 token buffer)
  __bf16* blob  = (__bf16*)(ws);
  float*  stats = (float*)(ws + 0x60000);
  float*  musig = stats + 32;
  __bf16* x_in  = (__bf16*)(ws + 0x80000);
  __bf16* z     = (__bf16*)(ws + 0x80000 + 1 * BUF);
  __bf16* u     = (__bf16*)(ws + 0x80000 + 2 * BUF);
  __bf16* delta = (__bf16*)(ws + 0x80000 + 3 * BUF);
  float*  BC    = (float*) (ws + 0x80000 + 4 * BUF);   // NTOK*32*4 = 8MB
  __bf16* y     = x_in;            // x_in dead after conv
  float*  outb  = (float*)delta;   // delta dead after scan (same 32MB)
  float*  o     = (float*)d_out;

  k_convert  <<<(NBLOB + 255) / 256, 256, 0, stream>>>(mlp_w1, mlp_w2, in_w, xp_w,
                                                       dt_w, out_w, blob, stats);
  k_mlp_inproj<<<NTOK / 64, 128, 0, stream>>>(x, blob, mlp_b1, mlp_b2, x_in, z);
  k_conv     <<<NTOK, 256, 0, stream>>>(x_in, conv_w, conv_b, u);
  k_xproj    <<<NTOK / 64, 128, 0, stream>>>(u, blob, dt_b, BC, delta);
  k_scan     <<<NB, 256, 0, stream>>>(delta, u, BC, z, A_log, Dw, y);
  k_outproj  <<<NTOK / 64, 128, 0, stream>>>(y, blob, outb, stats);
  k_gnstats  <<<1, 16, 0, stream>>>(stats, musig);
  k_final    <<<(NB * D_MODEL * LSEQ) / 256, 256, 0, stream>>>(outb, musig, gn_w,
                                                               gn_b, x, o);
}
